// PrimaryUserPresenceNetwork_8589934592401
// MI455X (gfx1250) — matrix operand order repaired and checked by reading, compiled, not run on hardware
//
#include <hip/hip_runtime.h>
#include <hip/hip_bf16.h>

// Problem sizes (fixed by reference)
#define N_MOD 64
#define HSZ   512
#define TLEN  1024
#define G4H   2048   // 4*H

typedef __attribute__((ext_vector_type(16))) __bf16 v16bf;
typedef __attribute__((ext_vector_type(8)))  __bf16 v8bf;
typedef __attribute__((ext_vector_type(8)))  float  v8f;

__device__ __forceinline__ unsigned short f32_to_bf16_rne(float f) {
  unsigned int u = __float_as_uint(f);
  u += 0x7FFFu + ((u >> 16) & 1u);       // round to nearest even
  return (unsigned short)(u >> 16);
}

__device__ __forceinline__ float sigmoidf_fast(float x) {
  return 1.0f / (1.0f + __expf(-x));
}

// ---------------------------------------------------------------------------
// Kernel 1: one-time W_hh fp32 -> bf16 conversion into workspace.
// 128 MB bf16 result stays resident in the 192 MB L2 for the whole recurrence.
// ---------------------------------------------------------------------------
__global__ void convert_whh_bf16(const float* __restrict__ whh,
                                 unsigned short* __restrict__ wbf,
                                 int n_elts) {
  int i = (blockIdx.x * blockDim.x + threadIdx.x) * 4;
  if (i + 3 < n_elts) {
    const float4 v = *(const float4*)(whh + i);
    wbf[i + 0] = f32_to_bf16_rne(v.x);
    wbf[i + 1] = f32_to_bf16_rne(v.y);
    wbf[i + 2] = f32_to_bf16_rne(v.z);
    wbf[i + 3] = f32_to_bf16_rne(v.w);
  }
}

// ---------------------------------------------------------------------------
// Kernel 2: one workgroup per LSTM module. 512 threads = 16 wave32.
// Per timestep: gates[2048] = W_hh_bf16 @ h_bf16 via chained
// v_wmma_f32_16x16x32_bf16 (B = h chunk broadcast to all 16 columns).
// A fragments for all 8 row-tiles of a K-chunk are staged into distinct
// registers first so the 16 global_load_b128 issue as one clause and the
// 8 independent WMMAs run back-to-back while the next chunk's loads fly.
// ---------------------------------------------------------------------------
__launch_bounds__(512)
__global__ void lstm_module_kernel(
    const float* __restrict__ recent,        // [64]
    const float* __restrict__ history,       // [64,1024]
    const float* __restrict__ W_ih,          // [64,2048]
    const float* __restrict__ b_ih,          // [64,2048]
    const float* __restrict__ b_hh,          // [64,2048]
    const float* __restrict__ lin_W,         // [64,513]
    const float* __restrict__ lin_b,         // [64]
    const unsigned short* __restrict__ Wbf,  // [64,2048,512] bf16
    float* __restrict__ per_mod)             // [64]
{
  __shared__ unsigned short h_bf[HSZ];   // current hidden state, bf16
  __shared__ float gates[G4H];           // raw recurrent gates (also reduce buf)
  __shared__ float h_f32[HSZ];           // final-step hidden, fp32

  const int n    = blockIdx.x;
  const int tid  = threadIdx.x;          // 0..511 ; j = tid owns h[j], c[j]
  const int lane = tid & 31;
  const int wave = tid >> 5;             // 0..15, owns gate rows [128w,128w+128)

  // per-thread input-projection params for rows j, 512+j, 1024+j, 1536+j
  const int j = tid;
  const float wih_i = W_ih[n * G4H + j];
  const float wih_f = W_ih[n * G4H + HSZ + j];
  const float wih_g = W_ih[n * G4H + 2 * HSZ + j];
  const float wih_o = W_ih[n * G4H + 3 * HSZ + j];
  const float bs_i = b_ih[n * G4H + j]           + b_hh[n * G4H + j];
  const float bs_f = b_ih[n * G4H + HSZ + j]     + b_hh[n * G4H + HSZ + j];
  const float bs_g = b_ih[n * G4H + 2 * HSZ + j] + b_hh[n * G4H + 2 * HSZ + j];
  const float bs_o = b_ih[n * G4H + 3 * HSZ + j] + b_hh[n * G4H + 3 * HSZ + j];

  float c = 0.0f;
  h_bf[j] = 0;                            // h0 = 0
  __syncthreads();

  // Base pointer for this wave's 128-row slab of the module's bf16 W_hh,
  // already offset by this lane's row-within-tile and K-half position.
  const int halfsel = (lane >> 4) & 1;    // 0: lanes 0-15, 1: lanes 16-31
  const int mlane   = lane & 15;
  const unsigned short* Wlane =
      Wbf + (size_t)n * G4H * HSZ
          + (size_t)(wave * 128 + mlane) * HSZ
          + halfsel * 8;

  const v8f vzero = {0.f, 0.f, 0.f, 0.f, 0.f, 0.f, 0.f, 0.f};

  for (int t = 0; t < TLEN; ++t) {
    // -------- recurrent matvec: 8 row-tiles of 16, K = 512 in 16 chunks ----
    v8f acc[8];
    #pragma unroll
    for (int tt = 0; tt < 8; ++tt) acc[tt] = vzero;

    #pragma unroll 2
    for (int kc = 0; kc < 16; ++kc) {
      const int k0 = kc * 32;
      // B fragment: h[k0..k0+31] broadcast into all 16 columns.
      // Lanes 0-15 hold K=k0+[0..7],[16..23]; lanes 16-31 hold K=k0+[8..15],[24..31].
      const unsigned short* hp = &h_bf[k0 + halfsel * 8];
      v8bf b_lo = *(const v8bf*)(hp);
      v8bf b_hi = *(const v8bf*)(hp + 16);
      v16bf bfrag = __builtin_shufflevector(
          b_lo, b_hi, 0, 1, 2, 3, 4, 5, 6, 7, 8, 9, 10, 11, 12, 13, 14, 15);

      // Stage A fragments for all 8 tiles (distinct registers -> 16 loads
      // in flight before the first WMMA needs data).
      v16bf av[8];
      #pragma unroll
      for (int tt = 0; tt < 8; ++tt) {
        // A (16x32 bf16): lane<16 -> M=row0+lane, K=[0..7],[16..23];
        // lane>=16 -> M=row0+lane-16, K=[8..15],[24..31].
        const unsigned short* ap = Wlane + (size_t)tt * 16 * HSZ + k0;
        v8bf a_lo = *(const v8bf*)(ap);        // K = off .. off+7
        v8bf a_hi = *(const v8bf*)(ap + 16);   // K = off+16 .. off+23
        av[tt] = __builtin_shufflevector(
            a_lo, a_hi, 0, 1, 2, 3, 4, 5, 6, 7, 8, 9, 10, 11, 12, 13, 14, 15);
      }
      // 8 independent WMMAs (disjoint accumulators, no RAW hazards).
      #pragma unroll
      for (int tt = 0; tt < 8; ++tt) {
        acc[tt] = __builtin_amdgcn_wmma_f32_16x16x32_bf16(
            false, av[tt], false, bfrag, (short)0, acc[tt], false, false);
      }
    }

    // D columns are identical; lanes 0 and 16 hold rows [0..7] / [8..15].
    if (mlane == 0) {
      #pragma unroll
      for (int tt = 0; tt < 8; ++tt) {
        const int rowbase = wave * 128 + tt * 16 + halfsel * 8;
        #pragma unroll
        for (int r = 0; r < 8; ++r) gates[rowbase + r] = acc[tt][r];
      }
    }
    __syncthreads();

    // -------- gate nonlinearity + state update (thread j owns h[j],c[j]) ---
    const float x = history[n * TLEN + t];   // uniform scalar -> s_load
    const float gi = gates[j]            + x * wih_i + bs_i;
    const float gf = gates[HSZ + j]      + x * wih_f + bs_f;
    const float gg = gates[2 * HSZ + j]  + x * wih_g + bs_g;
    const float go = gates[3 * HSZ + j]  + x * wih_o + bs_o;
    c = sigmoidf_fast(gf) * c + sigmoidf_fast(gi) * tanhf(gg);
    const float h = sigmoidf_fast(go) * tanhf(c);
    h_bf[j] = f32_to_bf16_rne(h);
    if (t == TLEN - 1) h_f32[j] = h;
    __syncthreads();
  }

  // -------- per-module output linear: dot(h, lin_W[:512]) + tail -----------
  gates[j] = h_f32[j] * lin_W[n * (HSZ + 1) + j];
  __syncthreads();
  #pragma unroll
  for (int s = 256; s > 0; s >>= 1) {
    if (j < s) gates[j] += gates[j + s];
    __syncthreads();
  }
  if (j == 0) {
    per_mod[n] = gates[0]
               + recent[n] * lin_W[n * (HSZ + 1) + HSZ]
               + lin_b[n];
  }
}

// ---------------------------------------------------------------------------
// Kernel 3: final Linear(N -> 1)
// ---------------------------------------------------------------------------
__global__ void final_linear_kernel(const float* __restrict__ per_mod,
                                    const float* __restrict__ final_W,
                                    const float* __restrict__ final_b,
                                    float* __restrict__ out) {
  __shared__ float s[N_MOD];
  const int t = threadIdx.x;
  s[t] = per_mod[t] * final_W[t];
  __syncthreads();
  if (t == 0) {
    float a = final_b[0];
    #pragma unroll
    for (int k = 0; k < N_MOD; ++k) a += s[k];
    out[0] = a;
  }
}

// ---------------------------------------------------------------------------
extern "C" void kernel_launch(void* const* d_in, const int* in_sizes, int n_in,
                              void* d_out, int out_size, void* d_ws, size_t ws_size,
                              hipStream_t stream) {
  const float* recent  = (const float*)d_in[0];
  const float* history = (const float*)d_in[1];
  const float* W_ih    = (const float*)d_in[2];
  const float* W_hh    = (const float*)d_in[3];
  const float* b_ih    = (const float*)d_in[4];
  const float* b_hh    = (const float*)d_in[5];
  const float* lin_W   = (const float*)d_in[6];
  const float* lin_b   = (const float*)d_in[7];
  const float* final_W = (const float*)d_in[8];
  const float* final_b = (const float*)d_in[9];

  // Workspace layout: [0, 128MB) bf16 W_hh ; then per_mod[64] floats.
  unsigned short* Wbf = (unsigned short*)d_ws;
  float* per_mod = (float*)((char*)d_ws + (size_t)N_MOD * G4H * HSZ * sizeof(unsigned short));

  const int n_elts = N_MOD * G4H * HSZ;            // 67,108,864
  convert_whh_bf16<<<n_elts / 4 / 256, 256, 0, stream>>>(W_hh, Wbf, n_elts);

  lstm_module_kernel<<<N_MOD, 512, 0, stream>>>(
      recent, history, W_ih, b_ih, b_hh, lin_W, lin_b, Wbf, per_mod);

  final_linear_kernel<<<1, N_MOD, 0, stream>>>(per_mod, final_W, final_b,
                                               (float*)d_out);
}
